// SlotAttention_9758165696697
// MI455X (gfx1250) — matrix operand order repaired and checked by reading, not compiled, not run on hardware
//
#include <hip/hip_runtime.h>
#include <hip/hip_bf16.h>

typedef _Float16 v16h __attribute__((ext_vector_type(16)));
typedef _Float16 v8h  __attribute__((ext_vector_type(8)));
typedef float    v8f  __attribute__((ext_vector_type(8)));

#define BS_TOTAL 131072   // B*S = 16*8192
#define SEQ 8192
#define D 256
#define NB 16             // batches
#define NSLOT 10
#define NPAD 16

// ---------------- prep: f16 weight copies, initial slots, zero Q pad ----------------
__global__ void prep_kernel(const float* __restrict__ kw, const float* __restrict__ vw,
                            const float* __restrict__ noise, const float* __restrict__ mu,
                            const float* __restrict__ ls,
                            _Float16* __restrict__ kw16, _Float16* __restrict__ vw16,
                            _Float16* __restrict__ Qf16, float* __restrict__ slots) {
    int idx = blockIdx.x * 256 + threadIdx.x;   // 65536 threads
    kw16[idx] = (_Float16)kw[idx];
    vw16[idx] = (_Float16)vw[idx];
    Qf16[idx] = (_Float16)0.0f;
    if (idx < NB * NSLOT * D) {
        int d = idx & 255;
        slots[idx] = mu[d] + __expf(ls[d]) * noise[idx];
    }
}

// ---------------- fused LayerNorm + K/V projection (WMMA) ----------------
// block: 256 threads (8 waves), 64 rows of features. Writes K f16 [s][d], V^T f16 [d][s].
__global__ void ln_kv_kernel(const float* __restrict__ feats,
                             const float* __restrict__ w1, const float* __restrict__ b1,
                             const _Float16* __restrict__ kw16, const float* __restrict__ kb,
                             const _Float16* __restrict__ vw16, const float* __restrict__ vb,
                             _Float16* __restrict__ Kf16, _Float16* __restrict__ Vt16) {
    __shared__ _Float16 tile[64][D];   // 32 KB
    const int lane = threadIdx.x & 31;
    const int wave = threadIdx.x >> 5;
    const long rowBase = (long)blockIdx.x * 64;

    float wv[8], bv[8];
#pragma unroll
    for (int j = 0; j < 8; ++j) { wv[j] = w1[lane + 32 * j]; bv[j] = b1[lane + 32 * j]; }

    // LayerNorm: one row per wave-iteration, butterfly reduction
    for (int rr = 0; rr < 8; ++rr) {
        int r = wave * 8 + rr;
        const float* rowp = feats + (rowBase + r) * D;
        float x[8], s = 0.f, s2 = 0.f;
#pragma unroll
        for (int j = 0; j < 8; ++j) { x[j] = rowp[lane + 32 * j]; s += x[j]; s2 += x[j] * x[j]; }
#pragma unroll
        for (int off = 16; off > 0; off >>= 1) { s += __shfl_xor(s, off); s2 += __shfl_xor(s2, off); }
        float m = s * (1.0f / 256.0f);
        float var = s2 * (1.0f / 256.0f) - m * m;
        float rstd = rsqrtf(var + 1e-5f);
#pragma unroll
        for (int j = 0; j < 8; ++j)
            tile[r][lane + 32 * j] = (_Float16)((x[j] - m) * rstd * wv[j] + bv[j]);
    }
    __syncthreads();

    // 128 output tiles (4 row-tiles x 16 col-tiles x {K,V}) over 8 waves
    for (int t = 0; t < 16; ++t) {
        int tid = wave * 16 + t;
        int mat = tid >> 6;           // 0 = K, 1 = V
        int rt  = (tid >> 4) & 3;
        int ct  = tid & 15;
        const _Float16* W   = mat ? vw16 : kw16;
        const float*    bias = mat ? vb : kb;
        int col = ct * 16 + (lane & 15);
        float bb = bias[col];
        v8f c;
#pragma unroll
        for (int i = 0; i < 8; ++i) c[i] = bb;
        int arow = rt * 16 + (lane & 15);
        const _Float16* wcol = W + (long)col * D;
#pragma unroll
        for (int kt = 0; kt < 8; ++kt) {
            int ka = kt * 32 + ((lane & 16) ? 8 : 0);
            v8h alo = *(const v8h*)&tile[arow][ka];
            v8h ahi = *(const v8h*)&tile[arow][ka + 16];
            v16h a;
#pragma unroll
            for (int i = 0; i < 8; ++i) { a[i] = alo[i]; a[i + 8] = ahi[i]; }
            int kbb = kt * 32 + ((lane & 16) ? 8 : 0);
            v8h blo = *(const v8h*)(wcol + kbb);
            v8h bhi = *(const v8h*)(wcol + kbb + 16);
            v16h bf;
#pragma unroll
            for (int i = 0; i < 8; ++i) { bf[i] = blo[i]; bf[i + 8] = bhi[i]; }
            c = __builtin_amdgcn_wmma_f32_16x16x32_f16(false, a, false, bf, (short)0, c, false, false);
        }
        long srow0 = rowBase + rt * 16 + ((lane & 16) ? 8 : 0);
        if (mat == 0) {
#pragma unroll
            for (int i = 0; i < 8; ++i) Kf16[(srow0 + i) * D + col] = (_Float16)c[i];
        } else {
            v8h hv;
#pragma unroll
            for (int i = 0; i < 8; ++i) hv[i] = (_Float16)c[i];
            *(v8h*)&Vt16[(long)col * BS_TOTAL + srow0] = hv;
        }
    }
}

// ---------------- slot LayerNorm + Q projection (scalar f32, tiny) ----------------
__global__ void slotq_kernel(const float* __restrict__ slots,
                             const float* __restrict__ w2, const float* __restrict__ b2,
                             const float* __restrict__ qw, const float* __restrict__ qb,
                             _Float16* __restrict__ Qf16) {
    __shared__ float sn[D];
    __shared__ float rs[8], rs2[8];
    int row = blockIdx.x;                  // 0..159
    int b = row / NSLOT, n = row % NSLOT;
    int t = threadIdx.x;
    int lane = t & 31, wave = t >> 5;
    float x = slots[row * D + t];
    float s = x, s2 = x * x;
#pragma unroll
    for (int off = 16; off > 0; off >>= 1) { s += __shfl_xor(s, off); s2 += __shfl_xor(s2, off); }
    if (lane == 0) { rs[wave] = s; rs2[wave] = s2; }
    __syncthreads();
    float S = 0.f, S2 = 0.f;
#pragma unroll
    for (int w = 0; w < 8; ++w) { S += rs[w]; S2 += rs2[w]; }
    float m = S * (1.0f / 256.0f);
    float var = S2 * (1.0f / 256.0f) - m * m;
    float rstd = rsqrtf(var + 1e-5f);
    sn[t] = (x - m) * rstd * w2[t] + b2[t];
    __syncthreads();
    const float* wr = qw + t * D;
    float acc = qb[t];
#pragma unroll 4
    for (int d = 0; d < D; ++d) acc += sn[d] * wr[d];
    Qf16[(b * NPAD + n) * D + t] = (_Float16)acc;
}

// ---------------- scores (WMMA) + softmax over slot axis + eps -> attn f16 ----------------
// grid (64, 16): x = s-chunk of 128, y = batch; 8 waves, one 16n x 16s tile each
__global__ void attn_kernel(const _Float16* __restrict__ Qf16,
                            const _Float16* __restrict__ Kf16,
                            _Float16* __restrict__ attn16) {
    int b = blockIdx.y;
    int lane = threadIdx.x & 31, wave = threadIdx.x >> 5;
    int sBase = blockIdx.x * 128 + wave * 16;
    int scol = sBase + (lane & 15);
    const _Float16* Qb = Qf16 + b * NPAD * D + (lane & 15) * D;       // row n = lane%16
    const _Float16* Kr = Kf16 + ((long)b * SEQ + scol) * D;           // column s
    v8f c;
#pragma unroll
    for (int i = 0; i < 8; ++i) c[i] = 0.0f;
#pragma unroll
    for (int kt = 0; kt < 8; ++kt) {
        int ka = kt * 32 + ((lane & 16) ? 8 : 0);
        v8h alo = *(const v8h*)(Qb + ka);
        v8h ahi = *(const v8h*)(Qb + ka + 16);
        v16h a;
#pragma unroll
        for (int i = 0; i < 8; ++i) { a[i] = alo[i]; a[i + 8] = ahi[i]; }
        int kbb = kt * 32 + ((lane & 16) ? 8 : 0);
        v8h blo = *(const v8h*)(Kr + kbb);
        v8h bhi = *(const v8h*)(Kr + kbb + 16);
        v16h bf;
#pragma unroll
        for (int i = 0; i < 8; ++i) { bf[i] = blo[i]; bf[i + 8] = bhi[i]; }
        c = __builtin_amdgcn_wmma_f32_16x16x32_f16(false, a, false, bf, (short)0, c, false, false);
    }
    // softmax over n (tile rows): lane holds rows 0-7 (lo half) or 8-15 (hi half) of column scol
    bool hi = (lane & 16) != 0;
    float m;
    if (!hi) {
        m = c[0];
#pragma unroll
        for (int i = 1; i < 8; ++i) m = fmaxf(m, c[i]);
    } else {
        m = fmaxf(c[0], c[1]);                      // only n=8,9 valid in hi half
    }
    m = fmaxf(m, __shfl_xor(m, 16));
    float e[8], sl = 0.f;
#pragma unroll
    for (int i = 0; i < 8; ++i) e[i] = expf(c[i] - m);
    if (!hi) {
#pragma unroll
        for (int i = 0; i < 8; ++i) sl += e[i];
    } else {
        sl = e[0] + e[1];
    }
    sl += __shfl_xor(sl, 16);
    float rinv = 1.0f / sl;
    int nb = hi ? 8 : 0;
#pragma unroll
    for (int i = 0; i < 8; ++i)
        attn16[((long)(b * NPAD + nb + i)) * SEQ + scol] = (_Float16)(e[i] * rinv + 1e-7f);
}

// ---------------- updates = attn @ V (WMMA, S split 8-way, f32 atomics) ----------------
// grid (16,16,8): batch, d-coltile, s-chunk; one wave per block
__global__ void upd_kernel(const _Float16* __restrict__ attn16,
                           const _Float16* __restrict__ Vt16,
                           float* __restrict__ updates) {
    int b = blockIdx.x, ct = blockIdx.y, sc = blockIdx.z;
    int lane = threadIdx.x;
    int sOff = sc * 1024;
    int col = ct * 16 + (lane & 15);                               // d
    const _Float16* ar = attn16 + (long)(b * NPAD + (lane & 15)) * SEQ;  // row n
    const _Float16* vc = Vt16 + (long)col * BS_TOTAL + (long)b * SEQ;    // column d
    v8f c;
#pragma unroll
    for (int i = 0; i < 8; ++i) c[i] = 0.0f;
#pragma unroll 4
    for (int kt = 0; kt < 32; ++kt) {
        int ka = sOff + kt * 32 + ((lane & 16) ? 8 : 0);
        v8h alo = *(const v8h*)(ar + ka);
        v8h ahi = *(const v8h*)(ar + ka + 16);
        v16h a;
#pragma unroll
        for (int i = 0; i < 8; ++i) { a[i] = alo[i]; a[i + 8] = ahi[i]; }
        int kbb = sOff + kt * 32 + ((lane & 16) ? 8 : 0);
        v8h blo = *(const v8h*)(vc + kbb);
        v8h bhi = *(const v8h*)(vc + kbb + 16);
        v16h bf;
#pragma unroll
        for (int i = 0; i < 8; ++i) { bf[i] = blo[i]; bf[i + 8] = bhi[i]; }
        c = __builtin_amdgcn_wmma_f32_16x16x32_f16(false, a, false, bf, (short)0, c, false, false);
    }
    int nb = (lane & 16) ? 8 : 0;
#pragma unroll
    for (int i = 0; i < 8; ++i)
        unsafeAtomicAdd(&updates[(long)(b * NPAD + nb + i) * D + col], c[i]);
}

// ---------------- GRU cell: slots = gru(x=slots_prev, h=updates) ----------------
__global__ void gru_kernel(const float* __restrict__ slots_prev,
                           const float* __restrict__ updates,
                           const float* __restrict__ wih, const float* __restrict__ whh,
                           const float* __restrict__ bih, const float* __restrict__ bhh,
                           float* __restrict__ out) {
    __shared__ float sp[D], up[D];
    int row = blockIdx.x;                 // 0..159
    int b = row / NSLOT, n = row % NSLOT;
    int t = threadIdx.x;
    sp[t] = slots_prev[row * D + t];
    up[t] = updates[(b * NPAD + n) * D + t];
    __syncthreads();
    float gx[3], gh[3];
#pragma unroll
    for (int g = 0; g < 3; ++g) {
        const float* wi = wih + (g * D + t) * D;
        const float* wh = whh + (g * D + t) * D;
        float a = bih[g * D + t], h = bhh[g * D + t];
#pragma unroll 4
        for (int d = 0; d < D; ++d) { a += sp[d] * wi[d]; h += up[d] * wh[d]; }
        gx[g] = a; gh[g] = h;
    }
    float r  = 1.0f / (1.0f + expf(-(gx[0] + gh[0])));
    float z  = 1.0f / (1.0f + expf(-(gx[1] + gh[1])));
    float nn = tanhf(gx[2] + r * gh[2]);
    out[row * D + t] = (1.0f - z) * nn + z * up[t];
}

extern "C" void kernel_launch(void* const* d_in, const int* in_sizes, int n_in,
                              void* d_out, int out_size, void* d_ws, size_t ws_size,
                              hipStream_t stream) {
    (void)in_sizes; (void)n_in; (void)out_size; (void)ws_size;
    const float* features = (const float*)d_in[0];
    const float* noise    = (const float*)d_in[1];
    const float* slot_mu  = (const float*)d_in[2];
    const float* slot_ls  = (const float*)d_in[3];
    const float* norm1_w  = (const float*)d_in[4];
    const float* norm1_b  = (const float*)d_in[5];
    const float* norm2_w  = (const float*)d_in[6];
    const float* norm2_b  = (const float*)d_in[7];
    const float* q_w      = (const float*)d_in[8];
    const float* q_b      = (const float*)d_in[9];
    const float* k_w      = (const float*)d_in[10];
    const float* k_b      = (const float*)d_in[11];
    const float* v_w      = (const float*)d_in[12];
    const float* v_b      = (const float*)d_in[13];
    const float* w_ih     = (const float*)d_in[14];
    const float* w_hh     = (const float*)d_in[15];
    const float* b_ih     = (const float*)d_in[16];
    const float* b_hh     = (const float*)d_in[17];

    char* ws = (char*)d_ws;
    size_t off = 0;
    _Float16* kw16  = (_Float16*)(ws + off); off += (size_t)D * D * 2;                 // 128 KB
    _Float16* vw16  = (_Float16*)(ws + off); off += (size_t)D * D * 2;                 // 128 KB
    _Float16* Kf16  = (_Float16*)(ws + off); off += (size_t)BS_TOTAL * D * 2;          // 64 MB
    _Float16* Vt16  = (_Float16*)(ws + off); off += (size_t)BS_TOTAL * D * 2;          // 64 MB
    _Float16* attn16= (_Float16*)(ws + off); off += (size_t)NB * NPAD * SEQ * 2;       // 4 MB
    _Float16* Qf16  = (_Float16*)(ws + off); off += (size_t)NB * NPAD * D * 2;         // 128 KB
    float* slots_a  = (float*)(ws + off); off += (size_t)NB * NSLOT * D * 4;
    float* slots_b  = (float*)(ws + off); off += (size_t)NB * NSLOT * D * 4;
    float* updates  = (float*)(ws + off); off += (size_t)NB * NPAD * D * 4;

    prep_kernel<<<256, 256, 0, stream>>>(k_w, v_w, noise, slot_mu, slot_ls,
                                         kw16, vw16, Qf16, slots_a);
    ln_kv_kernel<<<2048, 256, 0, stream>>>(features, norm1_w, norm1_b,
                                           kw16, k_b, vw16, v_b, Kf16, Vt16);

    float* bufs[2] = {slots_a, slots_b};
    float* cur = slots_a;
    for (int it = 0; it < 3; ++it) {
        float* nxt = (it == 2) ? (float*)d_out : bufs[(it + 1) & 1];
        slotq_kernel<<<160, 256, 0, stream>>>(cur, norm2_w, norm2_b, q_w, q_b, Qf16);
        attn_kernel<<<dim3(64, 16), 256, 0, stream>>>(Qf16, Kf16, attn16);
        hipMemsetAsync(updates, 0, (size_t)NB * NPAD * D * 4, stream);
        upd_kernel<<<dim3(16, 16, 8), 32, 0, stream>>>(attn16, Vt16, updates);
        gru_kernel<<<160, 256, 0, stream>>>(cur, updates, w_ih, w_hh, b_ih, b_hh, nxt);
        cur = nxt;
    }
}
